// VanillaMoELayer_32899449487925
// MI455X (gfx1250) — matrix operand order repaired and checked by reading, not compiled, not run on hardware
//
#include <hip/hip_runtime.h>

#define N_TOKENS  16384
#define D_MODEL   1024
#define D_FF      4096
#define N_EXPERTS 8
#define TILE_M    64                  // tokens per workgroup (4 row-tiles of 16)
#define NRT       (TILE_M / 16)       // row tiles
#define FF_CHUNK  256                 // ff columns per chunk (16 waves x 16 cols)
#define XSTRIDE   (D_MODEL + 8)       // 1032 halfs: row advance = 4 dwords mod 64 banks
#define HSTRIDE   (FF_CHUNK + 8)      // 264 halfs: row advance = 4 dwords mod 64 banks

typedef _Float16 v16h __attribute__((ext_vector_type(16)));
typedef _Float16 h8   __attribute__((ext_vector_type(8)));
typedef float    v8f  __attribute__((ext_vector_type(8)));

// ---------------------------------------------------------------- init
__global__ void init_counts(int* cnt) {
    if (threadIdx.x < N_EXPERTS) cnt[threadIdx.x] = 0;
}

// ---------------------------------------------------------------- router
__global__ __launch_bounds__(256) void router_kernel(
    const float* __restrict__ x, const float* __restrict__ wr,
    int* __restrict__ cnt, int* __restrict__ bucket, float* __restrict__ wgt)
{
    int t = blockIdx.x * blockDim.x + threadIdx.x;
    if (t >= N_TOKENS) return;

    const float4* xr = (const float4*)(x + (size_t)t * D_MODEL);
    float acc[N_EXPERTS];
#pragma unroll
    for (int e = 0; e < N_EXPERTS; ++e) acc[e] = 0.0f;

    for (int k4 = 0; k4 < D_MODEL / 4; ++k4) {
        float4 xv = xr[k4];
#pragma unroll
        for (int e = 0; e < N_EXPERTS; ++e) {
            float4 wv = ((const float4*)(wr + (size_t)e * D_MODEL))[k4];
            acc[e] += xv.x * wv.x + xv.y * wv.y + xv.z * wv.z + xv.w * wv.w;
        }
    }
    float lmax = acc[0]; int idx = 0;
#pragma unroll
    for (int e = 1; e < N_EXPERTS; ++e)
        if (acc[e] > lmax) { lmax = acc[e]; idx = e; }   // first-max => argmax semantics
    float s = 0.0f;
#pragma unroll
    for (int e = 0; e < N_EXPERTS; ++e) s += __expf(acc[e] - lmax);

    wgt[t] = 1.0f / s;                                   // max softmax prob
    int pos = atomicAdd(&cnt[idx], 1);
    bucket[idx * N_TOKENS + pos] = t;
}

// ---------------------------------------------------------------- fused MoE FFN
// 512 threads = 16 waves. Workgroup: 64 tokens of one expert.
// GEMM1: wave wv computes h cols [wv*16, wv*16+16) for all 4 row-tiles.
// GEMM2: wave wv owns output cols [wv*64, wv*64+64) (4 col-tiles x 4 row-tiles).
__global__ __launch_bounds__(512) void moe_ffn_kernel(
    const float* __restrict__ x,  const float* __restrict__ w1,
    const float* __restrict__ w2, const int* __restrict__ cnt,
    const int* __restrict__ bucket, const float* __restrict__ wgt,
    float* __restrict__ out)
{
    __shared__ _Float16 lds_x[TILE_M][XSTRIDE];   // ~132 KB
    __shared__ _Float16 lds_h[TILE_M][HSTRIDE];   // ~34 KB
    __shared__ int   s_tok[TILE_M];
    __shared__ float s_w[TILE_M];

    const int e    = blockIdx.x >> 8;         // 256 tiles per expert
    const int tile = blockIdx.x & 255;
    const int n    = cnt[e];
    const int base = tile * TILE_M;
    if (base >= n) return;

    const int tid  = threadIdx.x;
    const int lane = tid & 31;
    const int wv   = tid >> 5;                // wave 0..15
    const int row  = lane & 15;               // A-row / B-column select
    const int half = lane >> 4;               // K-half select
    const int koffA = half * 16;

    if (tid < TILE_M) {
        int r  = base + tid;
        int tk = bucket[e * N_TOKENS + ((r < n) ? r : (n - 1))];
        s_tok[tid] = tk;
        s_w[tid]   = (r < n) ? wgt[tk] : 0.0f;
    }
    __syncthreads();

    // stage x tile (f32 -> f16) into LDS
    for (int i = tid; i < TILE_M * D_MODEL; i += 512) {
        int r = i >> 10, c = i & (D_MODEL - 1);
        lds_x[r][c] = (_Float16)x[(size_t)s_tok[r] * D_MODEL + c];
    }
    __syncthreads();

    const float* w1e = w1 + (size_t)e * D_FF * D_MODEL;
    const float* w2e = w2 + (size_t)e * D_MODEL * D_FF;

    const v8f zero = {};
    v8f oacc[NRT][4];
#pragma unroll
    for (int rt = 0; rt < NRT; ++rt)
#pragma unroll
        for (int g = 0; g < 4; ++g) oacc[rt][g] = zero;

    for (int ff = 0; ff < D_FF; ff += FF_CHUNK) {
        // ---- GEMM1: h[:, wv*16 .. wv*16+16) over K = D_MODEL, all row tiles
        v8f hacc[NRT];
#pragma unroll
        for (int rt = 0; rt < NRT; ++rt) hacc[rt] = zero;

        const float* bptr = w1e + (size_t)(ff + wv * 16 + row) * D_MODEL + half * 16;
        for (int k0 = 0; k0 < D_MODEL; k0 += 32) {
            const float4* q = (const float4*)(bptr + k0);
            __builtin_prefetch(bptr + k0 + 128, 0, 1);   // global_prefetch_b8
            v16h b;
#pragma unroll
            for (int j = 0; j < 4; ++j) {
                float4 f = q[j];
                b[4*j+0] = (_Float16)f.x; b[4*j+1] = (_Float16)f.y;
                b[4*j+2] = (_Float16)f.z; b[4*j+3] = (_Float16)f.w;
            }
#pragma unroll
            for (int rt = 0; rt < NRT; ++rt) {
                h8 lo = *(const h8*)&lds_x[rt * 16 + row][k0 + koffA];
                h8 hi = *(const h8*)&lds_x[rt * 16 + row][k0 + 8 + koffA];
                v16h a = __builtin_shufflevector(lo, hi,
                            0,1,2,3,4,5,6,7,8,9,10,11,12,13,14,15);
                hacc[rt] = __builtin_amdgcn_wmma_f32_16x16x32_f16(
                               false, a, false, b, (short)0, hacc[rt], false, false);
            }
        }
        // SiLU (fast rcp) + store f16 h tiles to LDS
        // C/D layout: element i -> M = i + 8*half, N = row
#pragma unroll
        for (int rt = 0; rt < NRT; ++rt) {
#pragma unroll
            for (int i = 0; i < 8; ++i) {
                float v   = hacc[rt][i];
                float sig = __builtin_amdgcn_rcpf(1.0f + __expf(-v));
                lds_h[rt * 16 + i + half * 8][wv * 16 + row] = (_Float16)(v * sig);
            }
        }
        __syncthreads();

        // ---- GEMM2: wave owns output cols [wv*64, wv*64+64), K = FF_CHUNK
#pragma unroll
        for (int g = 0; g < 4; ++g) {
            const int dcol = wv * 64 + g * 16 + row;
            const float* b2 = w2e + (size_t)dcol * D_FF + ff + half * 16;
            for (int kk = 0; kk < FF_CHUNK / 32; ++kk) {
                const float4* q = (const float4*)(b2 + kk * 32);
                v16h b;
#pragma unroll
                for (int j = 0; j < 4; ++j) {
                    float4 f = q[j];
                    b[4*j+0] = (_Float16)f.x; b[4*j+1] = (_Float16)f.y;
                    b[4*j+2] = (_Float16)f.z; b[4*j+3] = (_Float16)f.w;
                }
#pragma unroll
                for (int rt = 0; rt < NRT; ++rt) {
                    h8 lo = *(const h8*)&lds_h[rt * 16 + row][kk * 32 + koffA];
                    h8 hi = *(const h8*)&lds_h[rt * 16 + row][kk * 32 + 8 + koffA];
                    v16h a = __builtin_shufflevector(lo, hi,
                                0,1,2,3,4,5,6,7,8,9,10,11,12,13,14,15);
                    oacc[rt][g] = __builtin_amdgcn_wmma_f32_16x16x32_f16(
                                      false, a, false, b, (short)0, oacc[rt][g], false, false);
                }
            }
        }
        __syncthreads();   // protect lds_h before next chunk's GEMM1 writes
    }

    // ---- scale by routing weight and scatter to out (each token written once)
#pragma unroll
    for (int rt = 0; rt < NRT; ++rt) {
#pragma unroll
        for (int i = 0; i < 8; ++i) {
            int m = rt * 16 + i + half * 8;
            if (base + m < n) {
                int   t     = s_tok[m];
                float scale = s_w[m];
                float* orow = out + (size_t)t * D_MODEL + wv * 64;
#pragma unroll
                for (int g = 0; g < 4; ++g)
                    orow[g * 16 + row] = oacc[rt][g][i] * scale;
            }
        }
    }
}

// ---------------------------------------------------------------- launch
extern "C" void kernel_launch(void* const* d_in, const int* in_sizes, int n_in,
                              void* d_out, int out_size, void* d_ws, size_t ws_size,
                              hipStream_t stream) {
    const float* x  = (const float*)d_in[0];
    const float* wr = (const float*)d_in[1];
    const float* w1 = (const float*)d_in[2];
    const float* w2 = (const float*)d_in[3];
    float* out = (float*)d_out;

    // workspace layout: counts(16 ints pad) | buckets(8*16384 int) | weights(16384 f32)
    int*   cnt    = (int*)d_ws;
    int*   bucket = cnt + 16;
    float* wgt    = (float*)(bucket + N_EXPERTS * N_TOKENS);

    init_counts<<<1, 32, 0, stream>>>(cnt);
    router_kernel<<<N_TOKENS / 256, 256, 0, stream>>>(x, wr, cnt, bucket, wgt);
    moe_ffn_kernel<<<N_EXPERTS * (N_TOKENS / TILE_M), 512, 0, stream>>>(
        x, w1, w2, cnt, bucket, wgt, out);
}
